// ChannelMambaBlock_28106265985452
// MI455X (gfx1250) — compile-verified, matrix operand order checked by reading, not run on hardware
//
#include <hip/hip_runtime.h>

// ---------------------------------------------------------------------------
// CDNA5 (gfx1250) fused ChannelMambaBlock.
// All GEMMs use v_wmma_f32_16x16x32_f16 (wave32 WMMA). Activations staged in
// LDS as f16; weights pre-converted to f16 in d_ws; B fragments loaded from
// global (L2-resident). Each wave drives TWO independent accumulator chains
// (paired M-tiles sharing one B fragment) to avoid WMMA->WMMA stalls.
// ---------------------------------------------------------------------------

typedef _Float16 f16;
typedef __attribute__((ext_vector_type(8)))  _Float16 v8h;
typedef __attribute__((ext_vector_type(16))) _Float16 v16h;
typedef __attribute__((ext_vector_type(8)))  float    v8f;

union Frag16 { v16h v; v8h h[2]; };

#define C_      192
#define M_      64          // tokens per workgroup
#define DTRANK_ 12
#define DSTATE_ 8
#define K_      4

// f16 weight workspace layout (element offsets into d_ws)
#define OFF_WIN    0                       // w_in      (384 x 192)
#define OFF_SSMIN  73728                   // ssm_in_w  (384 x 192)
#define OFF_XPROJ  147456                  // x_proj_w  (112 x 192)
#define OFF_DTW    168960                  // dt_w pad  (4 x 192 x 32), r>=12 zero
#define OFF_SSMOUT 193536                  // ssm_out_w (192 x 192)
#define OFF_WOUT   230400                  // w_out     (192 x 192)
#define W_TOTAL    267264

__device__ __forceinline__ float siluf(float x)     { return x / (1.0f + __expf(-x)); }
__device__ __forceinline__ float softplusf(float x) { return (x > 20.0f) ? x : __logf(1.0f + __expf(x)); }

// Load a 16x32 f16 WMMA A/B fragment from a row-major [rows][ldk] f16 array.
// Per ISA 7.12.2 (16-bit A 16x32): lanes 0-15 hold K=0..7 and K=16..23 of row
// (lane&15); lanes 16-31 hold K=8..15 and K=24..31. B (32x16) mirrors with N
// in place of M (row-major [N][K] storage == W as given, since GEMMs are x@W^T).
__device__ __forceinline__ v16h ld_frag(const f16* __restrict__ base, int ldk,
                                        int row0, int k0, int lane) {
  const int r  = row0 + (lane & 15);
  const int kh = (lane >> 4) << 3;            // 0 or 8
  Frag16 f;
  f.h[0] = *(const v8h*)(base + r * ldk + k0 + kh);
  f.h[1] = *(const v8h*)(base + r * ldk + k0 + 16 + kh);
  return f.v;
}

#define WMMA_F16(acc, a, b) \
  __builtin_amdgcn_wmma_f32_16x16x32_f16(false, (a), false, (b), (short)0, (acc), false, false)

// ---------------------------------------------------------------------------
// Weight conversion pre-pass: fp32 -> f16 into d_ws, with dt_w K-padded 12->32.
// ---------------------------------------------------------------------------
__global__ void convert_weights_k(const float* __restrict__ w_in,
                                  const float* __restrict__ ssm_in,
                                  const float* __restrict__ xproj,
                                  const float* __restrict__ dtw,
                                  const float* __restrict__ ssm_out,
                                  const float* __restrict__ w_out,
                                  f16* __restrict__ ws) {
  int i = blockIdx.x * blockDim.x + threadIdx.x;
  if (i >= W_TOTAL) return;
  if (i < OFF_SSMIN)        ws[i] = (f16)w_in[i];
  else if (i < OFF_XPROJ)   ws[i] = (f16)ssm_in[i - OFF_SSMIN];
  else if (i < OFF_DTW)     ws[i] = (f16)xproj[i - OFF_XPROJ];
  else if (i < OFF_SSMOUT) {
    int j = i - OFF_DTW, kc = j >> 5, r = j & 31;
    ws[i] = (f16)((r < DTRANK_) ? dtw[kc * DTRANK_ + r] : 0.0f);
  }
  else if (i < OFF_WOUT)    ws[i] = (f16)ssm_out[i - OFF_SSMOUT];
  else                      ws[i] = (f16)w_out[i - OFF_WOUT];
}

// ---------------------------------------------------------------------------
// Fused block kernel: one workgroup = 64 tokens (fixed b,h; 64 consecutive w).
// 256 threads = 8 wave32.
// ---------------------------------------------------------------------------
__global__ __launch_bounds__(256) void mamba_fused_k(
    const float* __restrict__ x,
    const float* __restrict__ norm_w,  const float* __restrict__ norm_b,
    const float* __restrict__ dt_b,    const float* __restrict__ Ds,
    const float* __restrict__ onorm_w, const float* __restrict__ onorm_b,
    const f16*   __restrict__ wh,
    float*       __restrict__ out) {

  __shared__ __attribute__((aligned(16))) f16 sh_xn [M_ * C_];     // xn -> (alias) y
  __shared__ __attribute__((aligned(16))) f16 sh_x1 [M_ * C_];     // x1 -> (alias) gemm6 input
  __shared__ __attribute__((aligned(16))) f16 sh_x2 [M_ * C_];
  __shared__ __attribute__((aligned(16))) f16 sh_z  [M_ * C_];
  __shared__ __attribute__((aligned(16))) f16 sh_xss[M_ * C_];
  __shared__ __attribute__((aligned(16))) f16 sh_dtr[M_ * (K_ * 32)]; // K-padded dtr
  __shared__ float sh_bvcv[M_ * K_ * 16];   // Bv (0..7) | Cv (8..15) per (m,k)
  __shared__ float sh_bc  [M_ * K_];
  __shared__ float sh_mu  [M_];
  __shared__ float sh_rs  [M_];

  f16* sh_y  = sh_xn;   // xn dead after GEMM1
  f16* sh_a6 = sh_x1;   // x1 dead after GEMM2

  const int tid  = threadIdx.x;
  const int lane = tid & 31;
  const int wave = tid >> 5;

  const int bid    = blockIdx.x;          // 1024 = 4(B) * 128(H) * 2(W/64)
  const int wchunk = bid & 1;
  const int h      = (bid >> 1) & 127;
  const int b      = bid >> 8;
  const int w0     = wchunk * M_;

  // Warm L2 with the f16 weights (global_prefetch_b8).
  if (tid < 128) __builtin_prefetch(wh + tid * 4096, 0, 1);

  // ---- Phase A: load x tile (coalesced over w) into LDS as f16; zero dtr pad.
  for (int i = tid; i < M_ * C_; i += 256) {
    int c = i / M_, m = i % M_;
    float v = x[((size_t)(b * C_ + c) * 128 + h) * 128 + (w0 + m)];
    sh_xn[m * C_ + c] = (f16)v;
  }
  for (int i = tid; i < M_ * (K_ * 32); i += 256) sh_dtr[i] = (f16)0.0f;
  __syncthreads();

  // ---- Phase B: LayerNorm #1 stats.
  if (tid < M_) {
    float s = 0.0f, s2 = 0.0f;
    for (int c = 0; c < C_; ++c) {
      float v = (float)sh_xn[tid * C_ + c];
      s += v; s2 += v * v;
    }
    float mu  = s * (1.0f / C_);
    float var = s2 * (1.0f / C_) - mu * mu;
    sh_mu[tid] = mu;
    sh_rs[tid] = rsqrtf(var + 1e-5f);
  }
  __syncthreads();
  for (int i = tid; i < M_ * C_; i += 256) {
    int m = i / C_, c = i - m * C_;
    float v = (float)sh_xn[i];
    sh_xn[i] = (f16)(((v - sh_mu[m]) * sh_rs[m]) * norm_w[c] + norm_b[c]);
  }
  __syncthreads();

  // ---- Phase C: GEMM1  p = xn @ w_in^T  -> x1 | x2   (M=64, N=384, K=192)
  // Dual M-tile per wave: one B fragment feeds two independent WMMA chains.
  for (int t = wave; t < 2 * 24; t += 8) {
    int mp = (t & 1) << 1;                // M-tile pair base: 0 or 2
    int nt = t >> 1;
    v8f acc0 = {}, acc1 = {};
    for (int kt = 0; kt < 6; ++kt) {
      v16h bb = ld_frag(wh + OFF_WIN, C_, nt * 16, kt * 32, lane);
      v16h a0 = ld_frag(sh_xn, C_, mp * 16, kt * 32, lane);
      v16h a1 = ld_frag(sh_xn, C_, (mp + 1) * 16, kt * 32, lane);
      acc0 = WMMA_F16(acc0, a0, bb);
      acc1 = WMMA_F16(acc1, a1, bb);
    }
    int n  = nt * 16 + (lane & 15);
    int mh = (lane >> 4) << 3;
    f16* dst = (n < C_) ? sh_x1 : sh_x2;
    int  nn  = (n < C_) ? n : n - C_;
#pragma unroll
    for (int r = 0; r < 8; ++r) {
      dst[(mp * 16 + mh + r) * C_ + nn]       = (f16)acc0[r];
      dst[((mp + 1) * 16 + mh + r) * C_ + nn] = (f16)acc1[r];
    }
  }
  __syncthreads();

  // ---- Phase D: GEMM2  u = x1 @ ssm_in^T -> silu -> xss | z
  for (int t = wave; t < 2 * 24; t += 8) {
    int mp = (t & 1) << 1;
    int nt = t >> 1;
    v8f acc0 = {}, acc1 = {};
    for (int kt = 0; kt < 6; ++kt) {
      v16h bb = ld_frag(wh + OFF_SSMIN, C_, nt * 16, kt * 32, lane);
      v16h a0 = ld_frag(sh_x1, C_, mp * 16, kt * 32, lane);
      v16h a1 = ld_frag(sh_x1, C_, (mp + 1) * 16, kt * 32, lane);
      acc0 = WMMA_F16(acc0, a0, bb);
      acc1 = WMMA_F16(acc1, a1, bb);
    }
    int n  = nt * 16 + (lane & 15);
    int mh = (lane >> 4) << 3;
    f16* dst = (n < C_) ? sh_xss : sh_z;
    int  nn  = (n < C_) ? n : n - C_;
#pragma unroll
    for (int r = 0; r < 8; ++r) {
      dst[(mp * 16 + mh + r) * C_ + nn]       = (f16)siluf(acc0[r]);
      dst[((mp + 1) * 16 + mh + r) * C_ + nn] = (f16)siluf(acc1[r]);
    }
  }
  __syncthreads();

  // ---- Phase E: GEMM3  xd = xss @ x_proj^T  (N=112 = 4k * 28)
  for (int t = wave; t < 2 * 7; t += 8) {
    int mp = (t & 1) << 1;
    int nt = t >> 1;
    v8f acc0 = {}, acc1 = {};
    for (int kt = 0; kt < 6; ++kt) {
      v16h bb = ld_frag(wh + OFF_XPROJ, C_, nt * 16, kt * 32, lane);
      v16h a0 = ld_frag(sh_xss, C_, mp * 16, kt * 32, lane);
      v16h a1 = ld_frag(sh_xss, C_, (mp + 1) * 16, kt * 32, lane);
      acc0 = WMMA_F16(acc0, a0, bb);
      acc1 = WMMA_F16(acc1, a1, bb);
    }
    int n  = nt * 16 + (lane & 15);       // 0..111
    int k  = n / 28, j = n - k * 28;
    int mh = (lane >> 4) << 3;
#pragma unroll
    for (int p = 0; p < 2; ++p) {
      int mb = (mp + p) * 16 + mh;
#pragma unroll
      for (int r = 0; r < 8; ++r) {
        int m = mb + r;
        float v = (p == 0) ? acc0[r] : acc1[r];
        if (j < DTRANK_)          sh_dtr[m * (K_ * 32) + k * 32 + j] = (f16)v;
        else if (j < DTRANK_ + 8) sh_bvcv[(m * K_ + k) * 16 + (j - DTRANK_)]         = v;
        else                      sh_bvcv[(m * K_ + k) * 16 + 8 + (j - DTRANK_ - 8)] = v;
      }
    }
  }
  __syncthreads();

  // ---- Phase F: bc[m][k] = sum_i Bv*Cv
  {
    int m = tid & 63, k = tid >> 6;       // 256 threads == 64*4 entries
    const float* p = &sh_bvcv[(m * K_ + k) * 16];
    float s = 0.0f;
#pragma unroll
    for (int i = 0; i < DSTATE_; ++i) s += p[i] * p[8 + i];
    sh_bc[m * K_ + k] = s;
  }
  __syncthreads();

  // ---- Phase G: GEMM4 per-k  dt = softplus(dtr @ dt_w^T + dt_b);
  //      gain = sum_k bc_k*dt_k + sum_k Ds ;  y = xss * gain
  // (4 per-k WMMAs are independent chains already, separated by softplus VALU.)
  for (int t = wave; t < 4 * 12; t += 8) {
    int mt = t & 3, nt = t >> 2;
    int n  = nt * 16 + (lane & 15);
    int mb = mt * 16 + ((lane >> 4) << 3);
    float dsum = Ds[n] + Ds[C_ + n] + Ds[2 * C_ + n] + Ds[3 * C_ + n];
    v8f g = {};
    for (int k = 0; k < K_; ++k) {
      v8f acc = {};
      v16h a = ld_frag(sh_dtr, K_ * 32, mt * 16, k * 32, lane);
      v16h bb = ld_frag(wh + OFF_DTW + k * C_ * 32, 32, nt * 16, 0, lane);
      acc = WMMA_F16(acc, a, bb);
      float bias = dt_b[k * C_ + n];
#pragma unroll
      for (int r = 0; r < 8; ++r) {
        float dtv = softplusf(acc[r] + bias);
        g[r] += sh_bc[(mb + r) * K_ + k] * dtv;
      }
    }
#pragma unroll
    for (int r = 0; r < 8; ++r) {
      int m = mb + r;
      float yv = (float)sh_xss[m * C_ + n] * (g[r] + dsum);
      sh_y[m * C_ + n] = (f16)yv;
    }
  }
  __syncthreads();

  // ---- Phase H: LayerNorm #2 on y, then * z.
  if (tid < M_) {
    float s = 0.0f, s2 = 0.0f;
    for (int c = 0; c < C_; ++c) {
      float v = (float)sh_y[tid * C_ + c];
      s += v; s2 += v * v;
    }
    float mu  = s * (1.0f / C_);
    float var = s2 * (1.0f / C_) - mu * mu;
    sh_mu[tid] = mu;
    sh_rs[tid] = rsqrtf(var + 1e-5f);
  }
  __syncthreads();
  for (int i = tid; i < M_ * C_; i += 256) {
    int m = i / C_, c = i - m * C_;
    float v  = (float)sh_y[i];
    float ln = ((v - sh_mu[m]) * sh_rs[m]) * onorm_w[c] + onorm_b[c];
    sh_y[i] = (f16)(ln * (float)sh_z[i]);
  }
  __syncthreads();

  // ---- Phase I: GEMM5  s = y @ ssm_out^T;  t = s * silu(x2)
  for (int t = wave; t < 2 * 12; t += 8) {
    int mp = (t & 1) << 1;
    int nt = t >> 1;
    v8f acc0 = {}, acc1 = {};
    for (int kt = 0; kt < 6; ++kt) {
      v16h bb = ld_frag(wh + OFF_SSMOUT, C_, nt * 16, kt * 32, lane);
      v16h a0 = ld_frag(sh_y, C_, mp * 16, kt * 32, lane);
      v16h a1 = ld_frag(sh_y, C_, (mp + 1) * 16, kt * 32, lane);
      acc0 = WMMA_F16(acc0, a0, bb);
      acc1 = WMMA_F16(acc1, a1, bb);
    }
    int n  = nt * 16 + (lane & 15);
    int mh = (lane >> 4) << 3;
#pragma unroll
    for (int r = 0; r < 8; ++r) {
      int m0 = mp * 16 + mh + r;
      int m1 = (mp + 1) * 16 + mh + r;
      sh_a6[m0 * C_ + n] = (f16)(acc0[r] * siluf((float)sh_x2[m0 * C_ + n]));
      sh_a6[m1 * C_ + n] = (f16)(acc1[r] * siluf((float)sh_x2[m1 * C_ + n]));
    }
  }
  __syncthreads();

  // ---- Phase J: GEMM6  o = t @ w_out^T;  out = x + o (transposed layout)
  for (int t = wave; t < 2 * 12; t += 8) {
    int mp = (t & 1) << 1;
    int nt = t >> 1;
    v8f acc0 = {}, acc1 = {};
    for (int kt = 0; kt < 6; ++kt) {
      v16h bb = ld_frag(wh + OFF_WOUT, C_, nt * 16, kt * 32, lane);
      v16h a0 = ld_frag(sh_a6, C_, mp * 16, kt * 32, lane);
      v16h a1 = ld_frag(sh_a6, C_, (mp + 1) * 16, kt * 32, lane);
      acc0 = WMMA_F16(acc0, a0, bb);
      acc1 = WMMA_F16(acc1, a1, bb);
    }
    int c  = nt * 16 + (lane & 15);
    int mh = (lane >> 4) << 3;
    size_t base = ((size_t)(b * C_ + c) * 128 + h) * 128 + w0;
#pragma unroll
    for (int r = 0; r < 8; ++r) {
      size_t i0 = base + (mp * 16 + mh + r);
      size_t i1 = base + ((mp + 1) * 16 + mh + r);
      out[i0] = acc0[r] + x[i0];
      out[i1] = acc1[r] + x[i1];
    }
  }
}

// ---------------------------------------------------------------------------
extern "C" void kernel_launch(void* const* d_in, const int* in_sizes, int n_in,
                              void* d_out, int out_size, void* d_ws, size_t ws_size,
                              hipStream_t stream) {
  const float* x        = (const float*)d_in[0];
  const float* norm_w   = (const float*)d_in[1];
  const float* norm_b   = (const float*)d_in[2];
  const float* w_in     = (const float*)d_in[3];
  const float* ssm_in_w = (const float*)d_in[4];
  const float* x_proj_w = (const float*)d_in[5];
  const float* dt_w     = (const float*)d_in[6];
  const float* dt_b     = (const float*)d_in[7];
  // d_in[8] = A_logs (unused by the reference computation)
  const float* Ds       = (const float*)d_in[9];
  const float* onorm_w  = (const float*)d_in[10];
  const float* onorm_b  = (const float*)d_in[11];
  const float* ssm_out  = (const float*)d_in[12];
  const float* w_out    = (const float*)d_in[13];

  f16*   wh  = (f16*)d_ws;
  float* out = (float*)d_out;

  convert_weights_k<<<(W_TOTAL + 255) / 256, 256, 0, stream>>>(
      w_in, ssm_in_w, x_proj_w, dt_w, ssm_out, w_out, wh);

  // grid: B(4) * H(128) * (W/64 = 2) = 1024 workgroups of 256 threads (8 waves)
  mamba_fused_k<<<1024, 256, 0, stream>>>(
      x, norm_w, norm_b, dt_b, Ds, onorm_w, onorm_b, wh, out);
}
